// Attention_91027536872049
// MI455X (gfx1250) — matrix operand order repaired and checked by reading, not compiled, not run on hardware
//
#include <hip/hip_runtime.h>
#include <hip/hip_bf16.h>
#include <math.h>

// ---------------------------------------------------------------------------
// Problem constants (match reference)
// ---------------------------------------------------------------------------
#define D_MODEL 2048
#define N_HEADS 32
#define D_KV    64
#define NUM_BUCKETS 32
#define MAX_DIST 128
#define BSZ 2
#define SSZ 2048
#define MROWS (BSZ * SSZ)    // 4096 rows for all big GEMMs
#define RELTAB (2 * SSZ - 1) // 4095

typedef __attribute__((ext_vector_type(16))) __bf16 v16bf;
typedef __attribute__((ext_vector_type(8)))  float  v8f;
typedef __attribute__((ext_vector_type(4)))  unsigned int u32x4;

union FragAB { v16bf v; u32x4 q[2]; __bf16 h[16]; };
union FragC  { v8f   v; float f[8]; };

// ---------------------------------------------------------------------------
// CDNA5 async memory->LDS helpers (ASYNCcnt-tracked DMA, no VGPR bounce)
// ---------------------------------------------------------------------------
__device__ __forceinline__ unsigned lds_off(const void* p) {
  // generic pointers to LDS are {aperture_hi32, lds_offset32}
  return (unsigned)(unsigned long long)(uintptr_t)p;
}

__device__ __forceinline__ void async_b128(unsigned ldsoff, const void* gptr) {
  asm volatile("global_load_async_to_lds_b128 %0, %1, off"
               :: "v"(ldsoff), "v"((unsigned long long)(uintptr_t)gptr)
               : "memory");
}

#define S_WAIT_ASYNC(n) asm volatile("s_wait_asynccnt %0" :: "n"(n) : "memory")

// ---------------------------------------------------------------------------
// fp32 -> bf16 conversion
// ---------------------------------------------------------------------------
__global__ __launch_bounds__(256)
void cvt_f32_bf16(const float* __restrict__ src, __bf16* __restrict__ dst, int n) {
  int i = (blockIdx.x * 256 + threadIdx.x) * 4;
  if (i + 3 < n) {
    dst[i + 0] = (__bf16)src[i + 0];
    dst[i + 1] = (__bf16)src[i + 1];
    dst[i + 2] = (__bf16)src[i + 2];
    dst[i + 3] = (__bf16)src[i + 3];
  }
}

// ---------------------------------------------------------------------------
// relative-position bucket table: bt[i] = bucket(rel = i-(S-1)), bidirectional
// ---------------------------------------------------------------------------
__global__ __launch_bounds__(256)
void bucket_kernel(int* __restrict__ bt) {
  int i = blockIdx.x * 256 + threadIdx.x;
  if (i >= RELTAB) return;
  int rel = i - (SSZ - 1);
  int nb = NUM_BUCKETS / 2;      // 16
  int max_exact = nb / 2;        // 8
  int b = (rel > 0) ? nb : 0;
  int ar = rel < 0 ? -rel : rel;
  if (ar < max_exact) {
    b += ar;
  } else {
    float lr = logf((float)ar / (float)max_exact) /
               logf((float)MAX_DIST / (float)max_exact) * (float)(nb - max_exact);
    int rl = max_exact + (int)lr;
    b += (rl < nb - 1 ? rl : nb - 1);
  }
  bt[i] = b;
}

// ---------------------------------------------------------------------------
// bf16 GEMM: C[M,N] = scale * A[M,K] x B[K,N]; A,B row-major bf16.
// Tile 128x128, BK=32, 256 threads = 8 waves; each wave: 2x4 grid of 16x16.
// Double-buffered LDS staged with global_load_async_to_lds_b128.
// ---------------------------------------------------------------------------
__global__ __launch_bounds__(256)
void gemm_bf16(const __bf16* __restrict__ A, const __bf16* __restrict__ Bm,
               void* __restrict__ Cout, int M, int N, int K,
               float scale, int out_bf16) {
  __shared__ __attribute__((aligned(16))) __bf16 As[2][128 * 40]; // pad 32->40
  __shared__ __attribute__((aligned(16))) __bf16 Bs[2][32 * 136]; // pad 128->136

  const int tid  = threadIdx.x;
  const int lane = tid & 31;
  const int wid  = tid >> 5;
  const int row0 = blockIdx.y * 128;
  const int col0 = blockIdx.x * 128;
  const int wm   = (wid >> 1) * 32;
  const int wn   = (wid & 1) * 64;

  FragC acc[2][4];
#pragma unroll
  for (int mt = 0; mt < 2; ++mt)
#pragma unroll
    for (int nt = 0; nt < 4; ++nt)
#pragma unroll
      for (int v = 0; v < 8; ++v) acc[mt][nt].f[v] = 0.0f;

  const int fm = lane & 15;
  const int kh = lane >> 4;

  const int idx = tid * 16;            // 256 threads x 16 bf16 = 4096 el/tile
  const int ra = idx >> 5, ca = idx & 31;    // A tile 128x32
  const int rb = idx >> 7, cb = idx & 127;   // B tile 32x128

  // async-stage one BK=32 slab into buffer `bf` (4 async issues per thread)
  auto stage = [&](int k0, int bf) {
    const __bf16* ga = A + (size_t)(row0 + ra) * K + k0 + ca;
    __bf16* la = &As[bf][ra * 40 + ca];
    async_b128(lds_off(la),     ga);
    async_b128(lds_off(la + 8), ga + 8);
    const __bf16* gb = Bm + (size_t)(k0 + rb) * N + col0 + cb;
    __bf16* lb = &Bs[bf][rb * 136 + cb];
    async_b128(lds_off(lb),     gb);
    async_b128(lds_off(lb + 8), gb + 8);
  };

  stage(0, 0);
  int buf = 0;
  for (int k0 = 0; k0 < K; k0 += 32) {
    if (k0 + 32 < K) {
      stage(k0 + 32, buf ^ 1);
      S_WAIT_ASYNC(4);   // 8 outstanding; oldest 4 (current slab) complete
    } else {
      S_WAIT_ASYNC(0);
    }
    __syncthreads();

    FragAB a[2], b[4];
#pragma unroll
    for (int mt = 0; mt < 2; ++mt) {
      const __bf16* p = &As[buf][(wm + mt * 16 + fm) * 40 + kh * 8];
      a[mt].q[0] = *(const u32x4*)p;
      a[mt].q[1] = *(const u32x4*)(p + 16);
    }
#pragma unroll
    for (int nt = 0; nt < 4; ++nt) {
      const __bf16* p = &Bs[buf][wn + nt * 16 + fm];
      for (int s = 0; s < 16; ++s) b[nt].h[s] = p[((s & 8) * 2 + kh * 8 + (s & 7)) * 136];
    }
#pragma unroll
    for (int mt = 0; mt < 2; ++mt)
#pragma unroll
      for (int nt = 0; nt < 4; ++nt)
        acc[mt][nt].v = __builtin_amdgcn_wmma_f32_16x16x32_bf16(
            false, a[mt].v, false, b[nt].v, (short)0, acc[mt][nt].v, false, false);
    __syncthreads();   // all reads done before next stage overwrites alt buf
    buf ^= 1;
  }

  const int er = lane >> 4;
  const int ec = lane & 15;
  if (out_bf16) {
    __bf16* O = (__bf16*)Cout;
#pragma unroll
    for (int mt = 0; mt < 2; ++mt)
#pragma unroll
      for (int nt = 0; nt < 4; ++nt)
#pragma unroll
        for (int v = 0; v < 8; ++v) {
          int r = row0 + wm + mt * 16 + v + 8 * er;
          int c = col0 + wn + nt * 16 + ec;
          O[(size_t)r * N + c] = (__bf16)(acc[mt][nt].f[v] * scale);
        }
  } else {
    float* O = (float*)Cout;
#pragma unroll
    for (int mt = 0; mt < 2; ++mt)
#pragma unroll
      for (int nt = 0; nt < 4; ++nt)
#pragma unroll
        for (int v = 0; v < 8; ++v) {
          int r = row0 + wm + mt * 16 + v + 8 * er;
          int c = col0 + wn + nt * 16 + ec;
          O[(size_t)r * N + c] = acc[mt][nt].f[v] * scale;
        }
  }
}

// ---------------------------------------------------------------------------
// Flash attention with relative-position bias.
// Grid: (head, batch, q-block of 64). 128 threads = 4 waves, one 16-q tile/wave.
// S^T = K*Q^T per 32-key chunk (q on the lane index -> softmax is one in-lane
// reduce + one shfl_xor(16)); P^T C-frags map directly onto PV A-fragments.
// K/V chunks async-staged once per workgroup into double-buffered LDS.
// ---------------------------------------------------------------------------
#define KVP 72   // LDS row pitch for 64-wide K/V rows (16B aligned, low conflict)

__global__ __launch_bounds__(128)
void attn_kernel(const __bf16* __restrict__ Qm, const __bf16* __restrict__ Km,
                 const __bf16* __restrict__ Vm, const float* __restrict__ mask,
                 const float* __restrict__ rel_emb, const int* __restrict__ bt,
                 __bf16* __restrict__ ctx) {
  __shared__ __attribute__((aligned(16))) float  bias_s[RELTAB];
  __shared__ __attribute__((aligned(16))) __bf16 qt_s[4][D_KV * 16];
  __shared__ __attribute__((aligned(16))) __bf16 kv_s[2][2][32 * KVP]; // [buf][K/V]

  const int tid  = threadIdx.x;
  const int lane = tid & 31;
  const int w    = tid >> 5;
  const int h    = blockIdx.x;
  const int b    = blockIdx.y;
  const int q0   = blockIdx.z * 64 + w * 16;

  const __bf16* Kbase = Km + (size_t)(b * SSZ) * D_MODEL + h * D_KV;
  const __bf16* Vbase = Vm + (size_t)(b * SSZ) * D_MODEL + h * D_KV;
  const float*  mb    = mask + (size_t)b * SSZ;

  // per-head bias table
  for (int i = tid; i < RELTAB; i += 128)
    bias_s[i] = rel_emb[bt[i] * N_HEADS + h];

  // build Q^T in LDS (one-time strided scatter)
  {
    int q = lane & 15, half = lane >> 4;
    const __bf16* gp = Qm + (size_t)(b * SSZ + q0 + q) * D_MODEL + h * D_KV + half * 32;
#pragma unroll
    for (int j = 0; j < 32; ++j)
      qt_s[w][(half * 32 + j) * 16 + q] = gp[j];
  }

  // async-stage a 32-key x 64-d K and V chunk (4 async issues per thread)
  const int sidx = tid * 16;               // 128 threads x 16 el = 2048 = 32x64
  const int srow = sidx >> 6, scol = sidx & 63;
  auto stage = [&](int kc, int bf) {
    const __bf16* kg = Kbase + (size_t)(kc + srow) * D_MODEL + scol;
    const __bf16* vg = Vbase + (size_t)(kc + srow) * D_MODEL + scol;
    __bf16* kl = &kv_s[bf][0][srow * KVP + scol];
    __bf16* vl = &kv_s[bf][1][srow * KVP + scol];
    async_b128(lds_off(kl),     kg);
    async_b128(lds_off(kl + 8), kg + 8);
    async_b128(lds_off(vl),     vg);
    async_b128(lds_off(vl + 8), vg + 8);
  };

  stage(0, 0);

  const int eq = lane & 15;   // q within tile (S^T N-index / PV A M-index)
  const int kh = lane >> 4;

  FragC o[4];
#pragma unroll
  for (int dn = 0; dn < 4; ++dn)
#pragma unroll
    for (int v = 0; v < 8; ++v) o[dn].f[v] = 0.0f;

  float mrun = -1e30f, lrun = 0.0f;

  int buf = 0;
  for (int kc = 0; kc < SSZ; kc += 32) {
    if (kc + 32 < SSZ) {
      stage(kc + 32, buf ^ 1);
      S_WAIT_ASYNC(4);
    } else {
      S_WAIT_ASYNC(0);
    }
    __syncthreads();

    const __bf16* Ks = kv_s[buf][0];
    const __bf16* Vs = kv_s[buf][1];

    // ---- S^T chunk: two 16(key)x16(q) C-frags, accumulated over d ----------
    FragC st[2];
#pragma unroll
    for (int kk = 0; kk < 2; ++kk) {
#pragma unroll
      for (int v = 0; v < 8; ++v) st[kk].f[v] = 0.0f;
#pragma unroll
      for (int dc = 0; dc < 2; ++dc) {
        FragAB ka, qb;
        const __bf16* kp = Ks + (kk * 16 + eq) * KVP + dc * 32 + kh * 8;
        ka.q[0] = *(const u32x4*)kp;
        ka.q[1] = *(const u32x4*)(kp + 16);
        const __bf16* qp = &qt_s[w][dc * 32 * 16 + eq];
        for (int s = 0; s < 16; ++s) qb.h[s] = qp[((s & 8) * 2 + kh * 8 + (s & 7)) * 16];
        st[kk].v = __builtin_amdgcn_wmma_f32_16x16x32_bf16(
            false, ka.v, false, qb.v, (short)0, st[kk].v, false, false);
      }
    }

    // ---- bias + mask + chunk max (row q == this lane) ----------------------
    float cmax = -1e30f;
#pragma unroll
    for (int kk = 0; kk < 2; ++kk)
#pragma unroll
      for (int v = 0; v < 8; ++v) {
        int key = kc + kk * 16 + v + 8 * kh;
        float s = st[kk].f[v] + bias_s[key - (q0 + eq) + (SSZ - 1)] + mb[key];
        st[kk].f[v] = s;
        cmax = fmaxf(cmax, s);
      }
    cmax = fmaxf(cmax, __shfl_xor(cmax, 16, 32));
    float mnew = fmaxf(mrun, cmax);
    float corr = __expf(mrun - mnew);
    mrun = mnew;

    // ---- P = exp(S - m), sum; pack straight into PV A-fragment -------------
    float csum = 0.0f;
    FragAB pa;
#pragma unroll
    for (int kk = 0; kk < 2; ++kk)
#pragma unroll
      for (int v = 0; v < 8; ++v) {
        float p = __expf(st[kk].f[v] - mnew);
        csum += p;
        pa.h[kk * 8 + v] = (__bf16)p;
      }
    csum += __shfl_xor(csum, 16, 32);
    lrun = lrun * corr + csum;

    // ---- rescale O (q on element index -> gather corr via bpermute) --------
    float corrv[8];
#pragma unroll
    for (int v = 0; v < 8; ++v) corrv[v] = __shfl(corr, v + 8 * kh, 32);
#pragma unroll
    for (int dn = 0; dn < 4; ++dn)
#pragma unroll
      for (int v = 0; v < 8; ++v) o[dn].f[v] *= corrv[v];

    // ---- O += P * V --------------------------------------------------------
#pragma unroll
    for (int dn = 0; dn < 4; ++dn) {
      FragAB vb;
      const __bf16* vp = Vs + dn * 16 + eq;
      for (int s = 0; s < 16; ++s) vb.h[s] = vp[((s & 8) * 2 + kh * 8 + (s & 7)) * KVP];
      o[dn].v = __builtin_amdgcn_wmma_f32_16x16x32_bf16(
          false, pa.v, false, vb.v, (short)0, o[dn].v, false, false);
    }
    __syncthreads();   // reads done before next stage overwrites alt buffer
    buf ^= 1;
  }

  // ---- finalize: divide by l, write ctx bf16 [b,s,h*64+d] ------------------
  float linv = 1.0f / lrun;
  float lv[8];
#pragma unroll
  for (int v = 0; v < 8; ++v) lv[v] = __shfl(linv, v + 8 * kh, 32);
#pragma unroll
  for (int dn = 0; dn < 4; ++dn)
#pragma unroll
    for (int v = 0; v < 8; ++v) {
      int r = b * SSZ + q0 + v + 8 * kh;
      int c = h * D_KV + dn * 16 + (lane & 15);
      ctx[(size_t)r * D_MODEL + c] = (__bf16)(o[dn].f[v] * lv[v]);
    }
}

// ---------------------------------------------------------------------------
// Host launcher
// ---------------------------------------------------------------------------
extern "C" void kernel_launch(void* const* d_in, const int* in_sizes, int n_in,
                              void* d_out, int out_size, void* d_ws, size_t ws_size,
                              hipStream_t stream) {
  const float* X    = (const float*)d_in[0];
  const float* mask = (const float*)d_in[1];
  const float* Wq   = (const float*)d_in[2];
  const float* Wk   = (const float*)d_in[3];
  const float* Wv   = (const float*)d_in[4];
  const float* Wo   = (const float*)d_in[5];
  const float* rel  = (const float*)d_in[6];

  char* p = (char*)d_ws;
  auto take = [&](size_t bytes) { char* r = p; p += (bytes + 255) & ~(size_t)255; return r; };
  const size_t NX = (size_t)MROWS * D_MODEL;
  const size_t NW = (size_t)D_MODEL * D_MODEL;

  __bf16* Xb  = (__bf16*)take(NX * 2);
  __bf16* Wqb = (__bf16*)take(NW * 2);
  __bf16* Wkb = (__bf16*)take(NW * 2);
  __bf16* Wvb = (__bf16*)take(NW * 2);
  __bf16* Wob = (__bf16*)take(NW * 2);
  __bf16* Qb  = (__bf16*)take(NX * 2);
  __bf16* Kb  = (__bf16*)take(NX * 2);
  __bf16* Vb  = (__bf16*)take(NX * 2);
  __bf16* Cb  = (__bf16*)take(NX * 2);
  int*    bt  = (int*)take(RELTAB * 4);

  cvt_f32_bf16<<<dim3((unsigned)(NX / 4 / 256)), 256, 0, stream>>>(X, Xb, (int)NX);
  cvt_f32_bf16<<<dim3((unsigned)(NW / 4 / 256)), 256, 0, stream>>>(Wq, Wqb, (int)NW);
  cvt_f32_bf16<<<dim3((unsigned)(NW / 4 / 256)), 256, 0, stream>>>(Wk, Wkb, (int)NW);
  cvt_f32_bf16<<<dim3((unsigned)(NW / 4 / 256)), 256, 0, stream>>>(Wv, Wvb, (int)NW);
  cvt_f32_bf16<<<dim3((unsigned)(NW / 4 / 256)), 256, 0, stream>>>(Wo, Wob, (int)NW);

  bucket_kernel<<<dim3((RELTAB + 255) / 256), 256, 0, stream>>>(bt);

  dim3 gg(D_MODEL / 128, MROWS / 128);  // 16 x 32
  gemm_bf16<<<gg, 256, 0, stream>>>(Xb, Wqb, Qb, MROWS, D_MODEL, D_MODEL, 0.125f, 1);
  gemm_bf16<<<gg, 256, 0, stream>>>(Xb, Wkb, Kb, MROWS, D_MODEL, D_MODEL, 1.0f, 1);
  gemm_bf16<<<gg, 256, 0, stream>>>(Xb, Wvb, Vb, MROWS, D_MODEL, D_MODEL, 1.0f, 1);

  attn_kernel<<<dim3(N_HEADS, BSZ, SSZ / 64), 128, 0, stream>>>(
      Qb, Kb, Vb, mask, rel, bt, Cb);

  gemm_bf16<<<gg, 256, 0, stream>>>(Cb, Wob, d_out, MROWS, D_MODEL, D_MODEL, 1.0f, 0);
}
